// Model_80839874445720
// MI455X (gfx1250) — matrix operand order repaired and checked by reading, not compiled, not run on hardware
//
#include <hip/hip_runtime.h>
#include <hip/hip_bf16.h>
#include <math.h>

typedef __attribute__((ext_vector_type(16))) _Float16 v16h;
typedef __attribute__((ext_vector_type(8)))  float    v8f;
typedef __attribute__((ext_vector_type(4)))  unsigned int v4u;
typedef __attribute__((ext_vector_type(8)))  int      v8i;
typedef __attribute__((ext_vector_type(4)))  int      v4i;

#define NNODE 65536
#define NEDGE 131072
#define NB    2048
#define NH    10
#define ND    32
#define HD    320      // NH*ND
#define GOUT  64
#define NODE_DIM 36
#define EDGE_DIM 40
#define CELL_IN  954
#define FMF   1024
#define FMK   32
#define FEAT  384
#define NPG   32
#define EPG   64

__device__ __forceinline__ float lrelu(float x) { return x > 0.f ? x : 0.2f * x; }
__device__ __forceinline__ float eluf(float x)  { return x > 0.f ? x : (__expf(x) - 1.f); }

// ---------------------------------------------------------------------------
// Generic WMMA GEMM: C[M,N] = act(A[M,K] @ B[K,N] + bias). f32 in/out, f16 MACs.
// One wave computes a 16x32 output tile (two 16x16 WMMA accs sharing one A frag).
// block = 256 threads = 8 waves. Main K loop is fully unchecked; one branchless
// masked remainder tile handles K % 32.
// A-frag (f16 16x32): lane<16 -> K {0..15}; lane>=16 -> {16..31} (K = 16*half + i)
// B-frag (32x16):     K = 16*half + i, N = lane&15
// ---------------------------------------------------------------------------
__global__ void gemm_f16_wmma(const float* __restrict__ A, int lda,
                              const float* __restrict__ Bm, int ldb,
                              const float* __restrict__ bias,
                              float* __restrict__ C, int ldc,
                              int M, int Nn, int K, int act)
{
  const int lane = threadIdx.x & 31;
  const int wav  = threadIdx.x >> 5;
  const int ntn  = Nn >> 5;                      // 32-wide column tiles
  const int wid  = blockIdx.x * 8 + wav;
  const int mt   = wid / ntn, nt = wid % ntn;
  if (mt * 16 >= M) return;                      // wave-uniform exit
  const int row0 = mt << 4;
  const int col0 = (nt << 5) + (lane & 15);
  const int half = lane >> 4;
  const int kaLo = half ? 16 : 0;
  const int kaHi = half ? 24 : 8;
  const int kb0  = half ? 16 : 0;
  const float* Arow = A + (size_t)(row0 + (lane & 15)) * lda;
  v8f acc0 = {}, acc1 = {};
  const int Kmain = K & ~31;
  int k0 = 0;
  for (; k0 < Kmain; k0 += 32) {
    if (k0 + 32 < K) __builtin_prefetch(&Arow[k0 + 32 + kaLo], 0, 1);
    v16h a, b0, b1;
    const float2* pLo = (const float2*)(Arow + k0 + kaLo);
    const float2* pHi = (const float2*)(Arow + k0 + kaHi);
#pragma unroll
    for (int i = 0; i < 4; ++i) {
      float2 lo = pLo[i], hi = pHi[i];
      a[2 * i]     = (_Float16)lo.x;
      a[2 * i + 1] = (_Float16)lo.y;
      a[8 + 2 * i]     = (_Float16)hi.x;
      a[8 + 2 * i + 1] = (_Float16)hi.y;
    }
#pragma unroll
    for (int i = 0; i < 16; ++i) {
      const float* Brow = Bm + (size_t)(k0 + kb0 + i) * ldb + col0;
      b0[i] = (_Float16)Brow[0];
      b1[i] = (_Float16)Brow[16];
    }
    acc0 = __builtin_amdgcn_wmma_f32_16x16x32_f16(false, a, false, b0, (short)0, acc0,
                                                  false, false);
    acc1 = __builtin_amdgcn_wmma_f32_16x16x32_f16(false, a, false, b1, (short)0, acc1,
                                                  false, false);
  }
  if (k0 < K) {   // branchless masked remainder tile
    v16h a, b0, b1;
#pragma unroll
    for (int i = 0; i < 8; ++i) {
      int k1 = k0 + kaLo + i;
      int k2 = k0 + kaHi + i;
      float v1 = Arow[k1 < K ? k1 : 0];
      float v2 = Arow[k2 < K ? k2 : 0];
      a[i]     = (_Float16)(k1 < K ? v1 : 0.f);
      a[8 + i] = (_Float16)(k2 < K ? v2 : 0.f);
    }
#pragma unroll
    for (int i = 0; i < 16; ++i) {
      int kk = k0 + kb0 + i;
      const float* Brow = Bm + (size_t)(kk < K ? kk : 0) * ldb + col0;
      float v0 = Brow[0];
      float v1 = Brow[16];
      b0[i] = (_Float16)(kk < K ? v0 : 0.f);
      b1[i] = (_Float16)(kk < K ? v1 : 0.f);
    }
    acc0 = __builtin_amdgcn_wmma_f32_16x16x32_f16(false, a, false, b0, (short)0, acc0,
                                                  false, false);
    acc1 = __builtin_amdgcn_wmma_f32_16x16x32_f16(false, a, false, b1, (short)0, acc1,
                                                  false, false);
  }
  float bb0 = bias ? bias[col0] : 0.f;
  float bb1 = bias ? bias[col0 + 16] : 0.f;
#pragma unroll
  for (int v = 0; v < 8; ++v) {
    int r = row0 + v + half * 8;
    float x0 = acc0[v] + bb0;
    float x1 = acc1[v] + bb1;
    if (act == 1) { x0 = fmaxf(x0, 0.f); x1 = fmaxf(x1, 0.f); }
    C[(size_t)r * ldc + col0]      = x0;
    C[(size_t)r * ldc + col0 + 16] = x1;
  }
}

// ---------------------------------------------------------------------------
// FM fused GEMM: per wave, one f (32 k-cols, two 16-wide WMMA tiles) x 16 rows.
// sumsq[b,f] = sum_k (X[b,:] @ Wfm[f,:,k])^2 computed in the epilogue via
// lane-half shfl_xor reduction over the C-layout columns.
// ---------------------------------------------------------------------------
__global__ void fm_sumsq_wmma(const float* __restrict__ X,
                              const float* __restrict__ Wfm,
                              float* __restrict__ sumsq)
{
  const int lane = threadIdx.x & 31;
  const int wav  = threadIdx.x >> 5;
  const int wid  = blockIdx.x * 8 + wav;     // 128 m-tiles * 1024 f
  const int f    = wid & (FMF - 1);
  const int mt   = wid >> 10;
  const int row0 = mt << 4;
  const int half = lane >> 4;
  const int kaLo = half ? 16 : 0;
  const int kaHi = half ? 24 : 8;
  const int kb0  = half ? 16 : 0;
  const int nlo  = lane & 15;
  const float* Arow = X + (size_t)(row0 + nlo) * FEAT;
  const float* Wf   = Wfm + (size_t)f * (FEAT * FMK);
  v8f a0 = {}, a1 = {};
  for (int k0 = 0; k0 < FEAT; k0 += 32) {
    if (k0 + 32 < FEAT) __builtin_prefetch(&Wf[(size_t)(k0 + 32 + kb0) * FMK + nlo], 0, 1);
    v16h a, b0, b1;
    const float2* pLo = (const float2*)(Arow + k0 + kaLo);
    const float2* pHi = (const float2*)(Arow + k0 + kaHi);
#pragma unroll
    for (int i = 0; i < 4; ++i) {
      float2 lo = pLo[i], hi = pHi[i];
      a[2 * i]     = (_Float16)lo.x;
      a[2 * i + 1] = (_Float16)lo.y;
      a[8 + 2 * i]     = (_Float16)hi.x;
      a[8 + 2 * i + 1] = (_Float16)hi.y;
    }
#pragma unroll
    for (int i = 0; i < 16; ++i) {
      const float* Brow = Wf + (size_t)(k0 + kb0 + i) * FMK + nlo;
      b0[i] = (_Float16)Brow[0];
      b1[i] = (_Float16)Brow[16];
    }
    a0 = __builtin_amdgcn_wmma_f32_16x16x32_f16(false, a, false, b0, (short)0, a0,
                                                false, false);
    a1 = __builtin_amdgcn_wmma_f32_16x16x32_f16(false, a, false, b1, (short)0, a1,
                                                false, false);
  }
#pragma unroll
  for (int v = 0; v < 8; ++v) {
    float s = a0[v] * a0[v] + a1[v] * a1[v];
    s += __shfl_xor(s, 1);
    s += __shfl_xor(s, 2);
    s += __shfl_xor(s, 4);
    s += __shfl_xor(s, 8);
    if (nlo == 0) sumsq[(size_t)(row0 + v + half * 8) * FMF + f] = s;
  }
}

// ---------------------------------------------------------------------------
// Folded attention vectors: u_s1[j,h] = sum_d W1[j,h*32+d]*a_s1[h,d], etc.
// ---------------------------------------------------------------------------
__global__ void prep_vecs(const float* __restrict__ W1, const float* __restrict__ a_s1,
                          const float* __restrict__ a_d1, const float* __restrict__ We1,
                          const float* __restrict__ a_e1, const float* __restrict__ We2,
                          const float* __restrict__ a_e2,
                          float* u_s1, float* u_d1, float* v_e1, float* v_e2)
{
  int id = blockIdx.x * blockDim.x + threadIdx.x;
  if (id < NODE_DIM * NH) {
    int j = id / NH, h = id % NH;
    float s = 0.f, d = 0.f;
    for (int dd = 0; dd < ND; ++dd) {
      float w = W1[j * HD + h * ND + dd];
      s += w * a_s1[h * ND + dd];
      d += w * a_d1[h * ND + dd];
    }
    u_s1[id] = s; u_d1[id] = d;
  } else if (id < NODE_DIM * NH + EDGE_DIM * NH) {
    int t = id - NODE_DIM * NH;
    int j = t / NH, h = t % NH;
    float s = 0.f;
    for (int dd = 0; dd < ND; ++dd) s += We1[j * HD + h * ND + dd] * a_e1[h * ND + dd];
    v_e1[t] = s;
  } else if (id < NODE_DIM * NH + EDGE_DIM * NH + EDGE_DIM) {
    int j = id - NODE_DIM * NH - EDGE_DIM * NH;
    float s = 0.f;
    for (int g = 0; g < GOUT; ++g) s += We2[j * GOUT + g] * a_e2[g];
    v_e2[j] = s;
  }
}

__global__ void node_alpha1(const float* __restrict__ x, const float* __restrict__ u_s1,
                            const float* __restrict__ u_d1, float* as1, float* ad1)
{
  int id = blockIdx.x * 256 + threadIdx.x;
  if (id >= NNODE * NH) return;
  int n = id / NH, h = id % NH;
  float s = 0.f, d = 0.f;
  for (int j = 0; j < NODE_DIM; ++j) {
    float xv = x[(size_t)n * NODE_DIM + j];
    s += xv * u_s1[j * NH + h];
    d += xv * u_d1[j * NH + h];
  }
  as1[id] = s; ad1[id] = d;
}

__global__ void edge_logit1(const float* __restrict__ ea, const float* __restrict__ v_e1,
                            float* el1)
{
  int id = blockIdx.x * 256 + threadIdx.x;
  if (id >= NEDGE * NH) return;
  int e = id / NH, h = id % NH;
  float s = 0.f;
  for (int j = 0; j < EDGE_DIM; ++j) s += ea[(size_t)e * EDGE_DIM + j] * v_e1[j * NH + h];
  el1[id] = s;
}

// One workgroup per molecule (32 nodes / 64 edges). Deterministic softmax
// aggregation entirely in LDS. block = 320 threads = (node 0..31) x (head 0..9).
// The 40KB xs tile (32 rows x 320 f32, contiguous) is staged into LDS by the
// Tensor Data Mover: wave 0 issues tensor_load_to_lds with a 1-D D# descriptor
// (data_size=4B, tile_dim0 = tensor_dim0 = 10240 elements), waits TENSORcnt=0,
// then the workgroup barrier releases the consumers.
__global__ void gat1_aggregate(const float* __restrict__ xs, const float* __restrict__ as1,
                               const float* __restrict__ ad1, const float* __restrict__ el1,
                               const int* __restrict__ src, const int* __restrict__ dst,
                               float* __restrict__ h1)
{
  __shared__ int   ssrc[EPG], sdst[EPG];
  __shared__ float slog[EPG][NH];
  __shared__ float swgt[EPG][NH];
  __shared__ float xsh[NPG][HD];
  int mol = blockIdx.x;
  int t   = threadIdx.x;   // 320
  int n0 = mol * NPG, e0 = mol * EPG;
  if (t < EPG) { ssrc[t] = src[e0 + t] - n0; sdst[t] = dst[e0 + t] - n0; }
  if (t < 32) {
    // --- TDM: DMA xs[n0*HD .. +10240) -> xsh, issued once by wave 0 ---
    const unsigned elems = NPG * HD;                       // 10240 x 4B = 40KB
    unsigned ldsa = (unsigned)(size_t)(&xsh[0][0]);        // low 32 bits of xsh's LDS address
    unsigned long long ga = (unsigned long long)(size_t)(xs + (size_t)n0 * HD);
    v4u g0;
    g0[0] = 1u;                                            // count=1, user descriptor
    g0[1] = ldsa;                                          // lds_addr
    g0[2] = (unsigned)(ga & 0xFFFFFFFFu);                  // global_addr[31:0]
    g0[3] = (unsigned)((ga >> 32) & 0x1FFFFFFu) | (2u << 30); // addr[56:32] | type=2
    v8i g1;
    g1[0] = (int)(2u << 16);                               // data_size = 4B
    g1[1] = (int)((elems & 0xFFFFu) << 16);                // tensor_dim0[15:0]
    g1[2] = (int)(((elems >> 16) & 0xFFFFu) | (1u << 16)); // dim0 hi | tensor_dim1=1
    g1[3] = (int)(elems << 16);                            // tile_dim0
    g1[4] = 0;                                             // tile_dim1/2 unused (1-D)
    g1[5] = (int)elems;                                    // tensor_dim0_stride lo32
    g1[6] = 0;
    g1[7] = 0;
    v4i z4 = {0, 0, 0, 0};
    v8i z8 = {0, 0, 0, 0, 0, 0, 0, 0};
    __builtin_amdgcn_tensor_load_to_lds(g0, g1, z4, z4, z8, 0);
    __builtin_amdgcn_s_wait_tensorcnt(0);
  }
  __syncthreads();
  for (int i = t; i < EPG * NH; i += 320) {
    int e = i / NH, h = i % NH;
    slog[e][h] = lrelu(as1[(size_t)(n0 + ssrc[e]) * NH + h] +
                       ad1[(size_t)(n0 + sdst[e]) * NH + h] +
                       el1[(size_t)(e0 + e) * NH + h]);
  }
  __syncthreads();
  int node = t & 31, h = t >> 5;   // h in 0..9
  float mx = -INFINITY;
  for (int e = 0; e < EPG; ++e) if (sdst[e] == node) mx = fmaxf(mx, slog[e][h]);
  float den = 1e-16f;
  for (int e = 0; e < EPG; ++e)
    if (sdst[e] == node) { float w = __expf(slog[e][h] - mx); swgt[e][h] = w; den += w; }
  float inv = 1.f / den;
  for (int d = 0; d < ND; ++d) {
    float acc = 0.f;
    for (int e = 0; e < EPG; ++e)
      if (sdst[e] == node) acc += swgt[e][h] * xsh[ssrc[e]][h * ND + d];
    h1[(size_t)(n0 + node) * HD + h * ND + d] = eluf(acc * inv);
  }
}

__global__ void node_alpha2(const float* __restrict__ y, const float* __restrict__ a_s2,
                            const float* __restrict__ a_d2, float* as2, float* ad2)
{
  int n = blockIdx.x * 256 + threadIdx.x;
  if (n >= NNODE) return;
  float s = 0.f, d = 0.f;
  for (int g = 0; g < GOUT; ++g) {
    float v = y[(size_t)n * GOUT + g];
    s += v * a_s2[g]; d += v * a_d2[g];
  }
  as2[n] = s; ad2[n] = d;
}

__global__ void edge_logit2(const float* __restrict__ ea, const float* __restrict__ v_e2,
                            float* el2)
{
  int e = blockIdx.x * 256 + threadIdx.x;
  if (e >= NEDGE) return;
  float s = 0.f;
  for (int j = 0; j < EDGE_DIM; ++j) s += ea[(size_t)e * EDGE_DIM + j] * v_e2[j];
  el2[e] = s;
}

__global__ void gat2_aggregate(const float* __restrict__ y, const float* __restrict__ as2,
                               const float* __restrict__ ad2, const float* __restrict__ el2,
                               const int* __restrict__ src, const int* __restrict__ dst,
                               float* __restrict__ h2)
{
  __shared__ int   ssrc[EPG], sdst[EPG];
  __shared__ float slog[EPG], swgt[EPG], sden[NPG];
  __shared__ float ysh[NPG][GOUT];
  int mol = blockIdx.x, t = threadIdx.x;  // 256
  int n0 = mol * NPG, e0 = mol * EPG;
  if (t < EPG) { ssrc[t] = src[e0 + t] - n0; sdst[t] = dst[e0 + t] - n0; }
  for (int i = t; i < NPG * GOUT; i += 256) ysh[i / GOUT][i % GOUT] = y[(size_t)n0 * GOUT + i];
  __syncthreads();
  if (t < EPG)
    slog[t] = lrelu(as2[n0 + ssrc[t]] + ad2[n0 + sdst[t]] + el2[e0 + t]);
  __syncthreads();
  if (t < NPG) {
    float mx = -INFINITY;
    for (int e = 0; e < EPG; ++e) if (sdst[e] == t) mx = fmaxf(mx, slog[e]);
    float den = 1e-16f;
    for (int e = 0; e < EPG; ++e)
      if (sdst[e] == t) { float w = __expf(slog[e] - mx); swgt[e] = w; den += w; }
    sden[t] = den;
  }
  __syncthreads();
  for (int i = t; i < NPG * GOUT; i += 256) {
    int node = i / GOUT, d = i % GOUT;
    float acc = 0.f;
    for (int e = 0; e < EPG; ++e) if (sdst[e] == node) acc += swgt[e] * ysh[ssrc[e]][d];
    h2[(size_t)n0 * GOUT + i] = eluf(acc / sden[node]);
  }
}

__global__ void readout(const float* __restrict__ h2, float* __restrict__ fall, int colofs)
{
  int mol = blockIdx.x;
  int d = threadIdx.x;   // 64
  float s = 0.f, mx = -INFINITY;
  for (int nn = 0; nn < NPG; ++nn) {
    float v = h2[(size_t)(mol * NPG + nn) * GOUT + d];
    s += v; mx = fmaxf(mx, v);
  }
  fall[(size_t)mol * FEAT + colofs + d] = s * (1.f / NPG);
  fall[(size_t)mol * FEAT + colofs + GOUT + d] = mx;
}

__global__ void l2norm_rows(float* __restrict__ x, int cols)
{
  __shared__ float sh[256];
  int row = blockIdx.x;
  float s = 0.f;
  for (int i = threadIdx.x; i < cols; i += 256) {
    float v = x[(size_t)row * cols + i];
    s += v * v;
  }
  sh[threadIdx.x] = s;
  __syncthreads();
  for (int o = 128; o > 0; o >>= 1) {
    if (threadIdx.x < o) sh[threadIdx.x] += sh[threadIdx.x + o];
    __syncthreads();
  }
  float inv = 1.f / fmaxf(sqrtf(sh[0]), 1e-12f);
  for (int i = threadIdx.x; i < cols; i += 256) x[(size_t)row * cols + i] *= inv;
}

__global__ void wfm_sq_t(const float* __restrict__ Wfm, float* __restrict__ W2sT)
{
  int id = blockIdx.x * 256 + threadIdx.x;
  if (id >= FEAT * FMF) return;
  int j = id / FMF, f = id % FMF;
  const float* p = Wfm + ((size_t)f * FEAT + j) * FMK;
  float s = 0.f;
  for (int k = 0; k < FMK; ++k) { float v = p[k]; s += v * v; }
  W2sT[id] = s;     // W2sT[j,f], row-major [384,1024]
}

__global__ void square_k(const float* __restrict__ x, float* __restrict__ y, int n)
{
  int id = blockIdx.x * 256 + threadIdx.x;
  if (id < n) { float v = x[id]; y[id] = v * v; }
}

__global__ void fm_combine(const float* __restrict__ lin, const float* __restrict__ sumsq,
                           const float* __restrict__ t2, float* __restrict__ fm, int n)
{
  int id = blockIdx.x * 256 + threadIdx.x;
  if (id < n) fm[id] = fmaxf(lin[id] + 0.5f * (sumsq[id] - t2[id]), 0.f);
}

__global__ void final_k(const float* __restrict__ z2, const float* __restrict__ Wk3,
                        const float* __restrict__ bk3, float* __restrict__ out)
{
  int b = blockIdx.x * 256 + threadIdx.x;
  if (b >= NB) return;
  float s = bk3[0];
  for (int g = 0; g < GOUT; ++g) s += z2[(size_t)b * GOUT + g] * Wk3[g];
  out[b] = 1.f / (1.f + __expf(-s));
}

// ---------------------------------------------------------------------------
extern "C" void kernel_launch(void* const* d_in, const int* in_sizes, int n_in,
                              void* d_out, int out_size, void* d_ws, size_t ws_size,
                              hipStream_t stream)
{
  (void)in_sizes; (void)n_in; (void)out_size; (void)ws_size;
  const float* x1   = (const float*)d_in[0];
  const int*   ei1  = (const int*)d_in[1];
  const float* ea1  = (const float*)d_in[2];
  const float* x2   = (const float*)d_in[4];
  const int*   ei2  = (const int*)d_in[5];
  const float* ea2  = (const float*)d_in[6];
  const float* cell = (const float*)d_in[8];
  const float* W1   = (const float*)d_in[9];
  const float* We1  = (const float*)d_in[10];
  const float* a_s1 = (const float*)d_in[11];
  const float* a_d1 = (const float*)d_in[12];
  const float* a_e1 = (const float*)d_in[13];
  const float* W2   = (const float*)d_in[14];
  const float* We2  = (const float*)d_in[15];
  const float* a_s2 = (const float*)d_in[16];
  const float* a_d2 = (const float*)d_in[17];
  const float* a_e2 = (const float*)d_in[18];
  const float* Wc1  = (const float*)d_in[19]; const float* bc1 = (const float*)d_in[20];
  const float* Wc2  = (const float*)d_in[21]; const float* bc2 = (const float*)d_in[22];
  const float* Wc3  = (const float*)d_in[23]; const float* bc3 = (const float*)d_in[24];
  const float* Wl   = (const float*)d_in[25]; const float* bl  = (const float*)d_in[26];
  const float* Wfm  = (const float*)d_in[27];
  const float* Wk1  = (const float*)d_in[28]; const float* bk1 = (const float*)d_in[29];
  const float* Wk2  = (const float*)d_in[30]; const float* bk2 = (const float*)d_in[31];
  const float* Wk3  = (const float*)d_in[32]; const float* bk3 = (const float*)d_in[33];
  float* out = (float*)d_out;

  // -------- workspace layout (floats) --------
  float* ws = (float*)d_ws;
  size_t o = 0;
  float* fall = ws + o;  o += (size_t)NB * FEAT;      // persistent concat features
  float* W2sT = ws + o;  o += (size_t)FEAT * FMF;     // sum_k Wfm^2, transposed
  float* u_s1 = ws + o;  o += 384;
  float* u_d1 = ws + o;  o += 384;
  float* v_e1 = ws + o;  o += 512;
  float* v_e2 = ws + o;  o += 64;
  float* A0   = ws + o;                               // big reusable region
  // GAT stage layout in region A
  float* xs   = A0;
  float* h1   = xs  + (size_t)NNODE * HD;
  float* y    = h1  + (size_t)NNODE * HD;
  float* h2   = y   + (size_t)NNODE * GOUT;
  float* as1b = h2  + (size_t)NNODE * GOUT;
  float* ad1b = as1b + (size_t)NNODE * NH;
  float* el1b = ad1b + (size_t)NNODE * NH;
  float* as2b = el1b + (size_t)NEDGE * NH;
  float* ad2b = as2b + NNODE;
  float* el2b = ad2b + NNODE;
  // dense stage reuses region A (GAT buffers dead by then)
  float* c1    = A0;
  float* c2    = c1   + (size_t)NB * 1024;
  float* lin   = c2   + (size_t)NB * 512;
  float* sumsq = lin  + (size_t)NB * FMF;
  float* term2 = sumsq + (size_t)NB * FMF;
  float* out2  = term2 + (size_t)NB * FMF;
  float* fm    = out2 + (size_t)NB * FEAT;
  float* z1    = fm   + (size_t)NB * FMF;
  float* z2    = z1   + (size_t)NB * 256;

  prep_vecs<<<4, 256, 0, stream>>>(W1, a_s1, a_d1, We1, a_e1, We2, a_e2,
                                   u_s1, u_d1, v_e1, v_e2);
  wfm_sq_t<<<(FEAT * FMF + 255) / 256, 256, 0, stream>>>(Wfm, W2sT);

  for (int side = 0; side < 2; ++side) {
    const float* x  = side ? x2 : x1;
    const int*   ei = side ? ei2 : ei1;
    const float* ea = side ? ea2 : ea1;
    const int* src = ei;
    const int* dst = ei + NEDGE;
    // xs = x @ W1  [65536,36]x[36,320]
    {
      int tiles = (NNODE / 16) * (HD / 32);
      gemm_f16_wmma<<<(tiles + 7) / 8, 256, 0, stream>>>(
          x, NODE_DIM, W1, HD, nullptr, xs, HD, NNODE, HD, NODE_DIM, 0);
    }
    node_alpha1<<<(NNODE * NH + 255) / 256, 256, 0, stream>>>(x, u_s1, u_d1, as1b, ad1b);
    edge_logit1<<<(NEDGE * NH + 255) / 256, 256, 0, stream>>>(ea, v_e1, el1b);
    gat1_aggregate<<<NB, 320, 0, stream>>>(xs, as1b, ad1b, el1b, src, dst, h1);
    // y = h1 @ W2  [65536,320]x[320,64]
    {
      int tiles = (NNODE / 16) * (GOUT / 32);
      gemm_f16_wmma<<<(tiles + 7) / 8, 256, 0, stream>>>(
          h1, HD, W2, GOUT, nullptr, y, GOUT, NNODE, GOUT, HD, 0);
    }
    node_alpha2<<<(NNODE + 255) / 256, 256, 0, stream>>>(y, a_s2, a_d2, as2b, ad2b);
    edge_logit2<<<(NEDGE + 255) / 256, 256, 0, stream>>>(ea, v_e2, el2b);
    gat2_aggregate<<<NB, 256, 0, stream>>>(y, as2b, ad2b, el2b, src, dst, h2);
    readout<<<NB, 64, 0, stream>>>(h2, fall, side * 128);
  }

  // ---- cell MLP (954 -> 1024 -> 512 -> 128, writes into fall cols 256..383) ----
  {
    int tiles = (NB / 16) * (1024 / 32);
    gemm_f16_wmma<<<(tiles + 7) / 8, 256, 0, stream>>>(
        cell, CELL_IN, Wc1, 1024, bc1, c1, 1024, NB, 1024, CELL_IN, 1);
  }
  {
    int tiles = (NB / 16) * (512 / 32);
    gemm_f16_wmma<<<(tiles + 7) / 8, 256, 0, stream>>>(
        c1, 1024, Wc2, 512, bc2, c2, 512, NB, 512, 1024, 1);
  }
  {
    int tiles = (NB / 16) * (128 / 32);
    gemm_f16_wmma<<<(tiles + 7) / 8, 256, 0, stream>>>(
        c2, 512, Wc3, 128, bc3, fall + 256, FEAT, NB, 128, 512, 0);
  }

  // ---- L2 norm of [B,384] concat ----
  l2norm_rows<<<NB, 256, 0, stream>>>(fall, FEAT);

  // ---- FM: lin, term2, fused sum_k t1^2, combine, norm ----
  {
    int tiles = (NB / 16) * (FMF / 32);
    gemm_f16_wmma<<<(tiles + 7) / 8, 256, 0, stream>>>(
        fall, FEAT, Wl, FMF, bl, lin, FMF, NB, FMF, FEAT, 0);
  }
  square_k<<<(NB * FEAT + 255) / 256, 256, 0, stream>>>(fall, out2, NB * FEAT);
  {
    int tiles = (NB / 16) * (FMF / 32);
    gemm_f16_wmma<<<(tiles + 7) / 8, 256, 0, stream>>>(
        out2, FEAT, W2sT, FMF, nullptr, term2, FMF, NB, FMF, FEAT, 0);
  }
  {
    int waves = (NB / 16) * FMF;      // 131072 waves
    fm_sumsq_wmma<<<waves / 8, 256, 0, stream>>>(fall, Wfm, sumsq);
  }
  fm_combine<<<(NB * FMF + 255) / 256, 256, 0, stream>>>(lin, sumsq, term2, fm, NB * FMF);
  l2norm_rows<<<NB, 256, 0, stream>>>(fm, FMF);

  // ---- classifier 1024 -> 256 -> 64 -> 1 ----
  {
    int tiles = (NB / 16) * (256 / 32);
    gemm_f16_wmma<<<(tiles + 7) / 8, 256, 0, stream>>>(
        fm, FMF, Wk1, 256, bk1, z1, 256, NB, 256, FMF, 1);
  }
  {
    int tiles = (NB / 16) * (64 / 32);
    gemm_f16_wmma<<<(tiles + 7) / 8, 256, 0, stream>>>(
        z1, 256, Wk2, 64, bk2, z2, 64, NB, 64, 256, 1);
  }
  final_k<<<(NB + 255) / 256, 256, 0, stream>>>(z2, Wk3, bk3, out);
}
